// ScaleDotProductAttention_69423851373088
// MI455X (gfx1250) — matrix operand order repaired and checked by reading, not compiled, not run on hardware
//
#include <hip/hip_runtime.h>
#include <hip/hip_bf16.h>

// Graph attention: per-edge QK^T (8x8 per edge) -> segment-mean -> softmax -> score*V.
// Edge phase uses V_WMMA_F32_16X16X32_BF16, two edges per wave packed block-diagonally.

typedef __attribute__((ext_vector_type(16))) __bf16        v16bf;
typedef __attribute__((ext_vector_type(8)))  float         v8f;
typedef __attribute__((ext_vector_type(4)))  unsigned int  v4u;

#define HH 8
#define DD 64

union Frag16 { v4u u[2]; v16bf v; };

__device__ __forceinline__ unsigned short f2bf(float f) {
    unsigned int u = __float_as_uint(f);
    unsigned int r = (u + 0x7FFFu + ((u >> 16) & 1u)) >> 16;   // RNE
    return (unsigned short)r;
}

// ---------------- zero workspace accumulators ----------------
__global__ __launch_bounds__(256)
void zero_kernel(float* __restrict__ p, int n) {
    int i = blockIdx.x * blockDim.x + threadIdx.x;
    if (i < n) p[i] = 0.0f;
}

// ---------------- fp32 -> bf16 streaming convert (vectorized) ----------------
__global__ __launch_bounds__(256)
void cvt_bf16_kernel(const float* __restrict__ src, unsigned short* __restrict__ dst, int n4) {
    int i = blockIdx.x * blockDim.x + threadIdx.x;
    if (i >= n4) return;
    float4 f = ((const float4*)src)[i];
    ushort4 o;
    o.x = f2bf(f.x); o.y = f2bf(f.y); o.z = f2bf(f.z); o.w = f2bf(f.w);
    ((ushort4*)dst)[i] = o;
}

// ---------------- per-edge scores via WMMA, atomic scatter ----------------
// One wave32 handles TWO edges. A-frag: rows 0-7 = q[dst0], rows 8-15 = q[dst1].
// B-frag: cols 0-7 = k[src0], cols 8-15 = k[src1]. Diagonal 8x8 blocks of C are
// the wanted scores; off-diagonal blocks are ignored.
__global__ __launch_bounds__(256)
void edge_score_kernel(const unsigned short* __restrict__ qb,   // [N,H,D] bf16
                       const unsigned short* __restrict__ kb,   // [N,H,D] bf16
                       const int* __restrict__ eidx,            // [2,E]
                       float* __restrict__ seg,                 // [N,8,8] f32
                       float* __restrict__ cnt,                 // [N] f32
                       int E) {
    const int lane = threadIdx.x & 31;
    const int wave = blockIdx.x * (blockDim.x >> 5) + (threadIdx.x >> 5);
    const int e0 = wave * 2;
    if (e0 >= E) return;                       // wave-uniform exit (EXEC stays full for WMMA)
    const bool has1 = (e0 + 1) < E;
    const int e1 = has1 ? (e0 + 1) : e0;

    const int dst0 = eidx[e0],     dst1 = eidx[e1];
    const int src0 = eidx[E + e0], src1 = eidx[E + e1];

    // ---- A fragment: 16x32 bf16 per K-step. Lane L holds row M = L&15.
    // Lanes 0-15: elements 0-7 -> K 0..7, elements 8-15 -> K 16..23.
    // Lanes 16-31: same +8 on K.  (ISA 7.12.2 16-bit A layout)
    const int M  = lane & 15;
    const int ah = M & 7;
    const long an = (long)((M < 8) ? dst0 : dst1);
    const unsigned short* arow = qb + (an * HH + ah) * DD;
    const int akoff = (lane < 16) ? 0 : 8;

    // ---- B fragment: 32x16 bf16. Lane L holds col N = L&15, same K order as A:
    // lanes 0-15 -> K 0..7,16..23; lanes 16-31 -> K 8..15,24..31.
    const int nn = lane & 15;
    const int bg = nn & 7;
    const long bn = (long)((nn < 8) ? src0 : src1);
    const unsigned short* brow = kb + (bn * HH + bg) * DD;
    const int bkoff = (lane < 16) ? 0 : 8;

    v8f c = {};
    #pragma unroll
    for (int t = 0; t < 2; ++t) {              // K = 64 in two 16x16x32 steps
        const int dbase = t * 32;
        Frag16 a, b;
        a.u[0] = *(const v4u*)(arow + dbase + akoff);
        a.u[1] = *(const v4u*)(arow + dbase + akoff + 16);
        b.u[0] = *(const v4u*)(brow + dbase + bkoff);
        b.u[1] = *(const v4u*)(brow + dbase + bkoff + 16);
        c = __builtin_amdgcn_wmma_f32_16x16x32_bf16(false, a.v, false, b.v,
                                                    (short)0, c, false, false);
    }

    // ---- scatter: C VGPR r = row h=r (lanes 0-15 -> M=r, 16-31 -> M=r+8).
    // edge0 block: lanes 0-7 (N=g=lane); edge1 block: lanes 24-31 (N=8..15 -> g=lane-24).
    const bool l0 = (lane < 8);
    const bool l1 = (lane >= 24) && has1;
    if (l0 | l1) {
        const int   g  = lane & 7;
        const long  dn = (long)(l0 ? dst0 : dst1);
        float* segp = seg + dn * 64 + g;
        #pragma unroll
        for (int h = 0; h < 8; ++h)
            unsafeAtomicAdd(segp + h * 8, c[h]);   // global_atomic_add_f32
    }
    if (lane == 0)          unsafeAtomicAdd(cnt + dst0, 1.0f);
    if (lane == 8 && has1)  unsafeAtomicAdd(cnt + dst1, 1.0f);
}

// ---------------- mean + softmax + score @ V ----------------
// 64 threads per node (thread j = feature d), 4 nodes per 256-thread block.
__global__ __launch_bounds__(256)
void finalize_kernel(const float* __restrict__ vmat,   // [N,H,D]
                     const float* __restrict__ seg,    // [N,8,8]
                     const float* __restrict__ cnt,    // [N]
                     float* __restrict__ out,          // [N,H,D]
                     int N) {
    __shared__ float s_raw[4 * 64];
    __shared__ float s_sc [4 * 64];
    const int ln = threadIdx.x >> 6;    // local node 0..3
    const int j  = threadIdx.x & 63;
    const int n  = blockIdx.x * 4 + ln;
    const bool ok = (n < N);

    const float cv  = ok ? cnt[n] : 1.0f;
    const float inv = 1.0f / fmaxf(cv, 1.0f);
    const float raw = ok ? seg[(long)n * 64 + j] * inv : 0.0f;
    s_raw[ln * 64 + j] = raw;
    __syncthreads();

    // softmax over g within row h = j/8
    const int h = j >> 3;
    float row[8];
    #pragma unroll
    for (int g = 0; g < 8; ++g) row[g] = s_raw[ln * 64 + h * 8 + g];
    float m = row[0];
    #pragma unroll
    for (int g = 1; g < 8; ++g) m = fmaxf(m, row[g]);
    float sum = 0.0f;
    #pragma unroll
    for (int g = 0; g < 8; ++g) sum += __expf(row[g] - m);
    s_sc[ln * 64 + j] = __expf(raw - m) / sum;
    __syncthreads();

    if (ok) {
        const int d = j;
        float vreg[8];
        #pragma unroll
        for (int g = 0; g < 8; ++g) vreg[g] = vmat[((long)n * 8 + g) * 64 + d];
        #pragma unroll
        for (int hh = 0; hh < 8; ++hh) {
            float acc = 0.0f;
            #pragma unroll
            for (int g = 0; g < 8; ++g) acc += s_sc[ln * 64 + hh * 8 + g] * vreg[g];
            out[((long)n * 8 + hh) * 64 + d] = acc;
        }
    }
}

extern "C" void kernel_launch(void* const* d_in, const int* in_sizes, int n_in,
                              void* d_out, int out_size, void* d_ws, size_t ws_size,
                              hipStream_t stream) {
    const float* q    = (const float*)d_in[0];
    const float* k    = (const float*)d_in[1];
    const float* v    = (const float*)d_in[2];
    const int*   eidx = (const int*)  d_in[3];

    const int NHD = in_sizes[0];              // N*H*D
    const int N   = NHD / (HH * DD);
    const int E   = in_sizes[3] / 2;
    float* out = (float*)d_out;

    // workspace: [seg N*64 f32][cnt N f32][pad][qb NHD bf16][kb NHD bf16]  (~115 MB)
    char* ws = (char*)d_ws;
    float* seg = (float*)ws;
    float* cnt = seg + (size_t)N * 64;
    size_t off = (((size_t)N * 65) * sizeof(float) + 31) & ~(size_t)31;
    unsigned short* qb = (unsigned short*)(ws + off);
    unsigned short* kb = qb + (size_t)NHD;

    // 1) zero accumulators (re-done every call: no cross-call state)
    {
        int nz = N * 65;
        zero_kernel<<<(nz + 255) / 256, 256, 0, stream>>>(seg, nz);
    }
    // 2) q,k -> bf16 (shrinks gather working set to ~102 MB: L2-resident)
    {
        int n4 = NHD / 4;
        cvt_bf16_kernel<<<(n4 + 255) / 256, 256, 0, stream>>>(q, qb, n4);
        cvt_bf16_kernel<<<(n4 + 255) / 256, 256, 0, stream>>>(k, kb, n4);
    }
    // 3) per-edge WMMA scores + atomic scatter (2 edges per wave)
    {
        int waves  = (E + 1) / 2;
        int blocks = (waves + 7) / 8;          // 8 waves (256 threads) per block
        edge_score_kernel<<<blocks, 256, 0, stream>>>(qb, kb, eidx, seg, cnt, E);
    }
    // 4) mean + softmax + score@V
    {
        int blocks = (N + 3) / 4;
        finalize_kernel<<<blocks, 256, 0, stream>>>(v, seg, cnt, out, N);
    }
}
